// ScaledDotProductAttention_34110630265664
// MI455X (gfx1250) — matrix operand order repaired and checked by reading, not compiled, not run on hardware
//
#include <hip/hip_runtime.h>
#include <hip/hip_bf16.h>

typedef __attribute__((ext_vector_type(16))) _Float16 v16h;
typedef __attribute__((ext_vector_type(8)))  _Float16 v8h;
typedef __attribute__((ext_vector_type(8)))  float    v8f;
typedef __attribute__((ext_vector_type(4)))  float    v4f;

#define SEQ    2048
#define DHEAD  64
#define NHEAD  16
#define NBATCH 4
#define NEGINF (-1.0e9f)

#define WMMA_F16(A, B, C) \
  __builtin_amdgcn_wmma_f32_16x16x32_f16(false, (A), false, (B), (short)0, (C), false, false)

// One wave32 per 16-row query tile of one (b,h).
// Causal skip: QK^T sweeps only cover key tiles kt <= qt; the fully-masked
// tail contributes the analytic constant p_tail = exp(-1e9 - m_row) per row
// (1.0 for fully-masked rows -> uniform softmax, matching the reference).
// Two-sweep softmax (max, then exp/sum with recomputed scores), f16 probs
// in 64 KB LDS, streamed out as normalized fp32 weights and fed to P*V WMMA.
__global__ __launch_bounds__(32) void attn_fused_wmma(
    const float* __restrict__ Q, const float* __restrict__ K,
    const float* __restrict__ V, const int* __restrict__ PadMask,
    float* __restrict__ Out, float* __restrict__ Attn)
{
  extern __shared__ char smem[];
  _Float16* probs  = (_Float16*)smem;                               // [16][SEQ]
  float*    rowsum = (float*)(smem + 16 * SEQ * sizeof(_Float16));  // [16]
  float*    rowmax = rowsum + 16;                                   // [16]

  const int lane = threadIdx.x;   // 0..31
  const int half = lane >> 4;     // 0 or 1
  const int l15  = lane & 15;

  const int qt = blockIdx.x & (SEQ / 16 - 1);
  const int bh = blockIdx.x >> 7;           // SEQ/16 == 128
  const int b  = bh / NHEAD;
  const int q0 = qt * 16;

  const float* qh = Q + (size_t)bh * SEQ * DHEAD;
  const float* kh = K + (size_t)bh * SEQ * DHEAD;
  const float* vh = V + (size_t)bh * SEQ * DHEAD;
  const int*   pm = PadMask + (size_t)b * SEQ;

  // ---- Q A-fragments (f32 -> f16). Contraction dims 0-31 -> a0, 32-63 -> a1.
  // A 16x32 f16 layout (ISA 7.12.2): lane row = l15; half h holds K
  // {8h..8h+7, 16+8h..16+8h+7}.
  v16h a0, a1;
  {
    const float* qrow = qh + (size_t)(q0 + l15) * DHEAD;
    const int kb = half * 8;
#pragma unroll
    for (int e = 0; e < 8; ++e) {
      a0[e]     = (_Float16)qrow[kb + e];
      a0[e + 8] = (_Float16)qrow[16 + kb + e];
      a1[e]     = (_Float16)qrow[32 + kb + e];
      a1[e + 8] = (_Float16)qrow[48 + kb + e];
    }
  }

  const int db = half * 8;      // B-fragment: lane col = l15, element e -> K = db+e+(e&8)
  const int ktiles = qt + 1;    // causal: only tiles [0, qt] have live columns

  auto loadKfrag = [&](int kt, v16h* b0, v16h* b1) {
    const float* krow = kh + (size_t)(kt * 16 + l15) * DHEAD;
#pragma unroll
    for (int e = 0; e < 16; ++e) {
      (*b0)[e] = (_Float16)krow[db + e + (e & 8)];
      (*b1)[e] = (_Float16)krow[32 + db + e + (e & 8)];
    }
  };

  // ================= Sweep 1: row max over live tiles =================
  float mrow[8];
#pragma unroll
  for (int i = 0; i < 8; ++i) mrow[i] = NEGINF;  // tail columns are all -1e9

  for (int kt = 0; kt < ktiles; ++kt) {
    if (kt + 1 < ktiles)
      __builtin_prefetch(kh + (size_t)(kt * 16 + 16 + l15) * DHEAD, 0, 3);
    v16h b0, b1;
    loadKfrag(kt, &b0, &b1);
    v8f c = {};
    c = WMMA_F16(a0, b0, c);
    c = WMMA_F16(a1, b1, c);
    const bool pmask = (pm[kt * 16 + l15] == 1);
    const bool diag  = (kt == qt);               // only tile crossing diagonal
#pragma unroll
    for (int i = 0; i < 8; ++i) {
      const int row = half * 8 + i;              // D layout: row = i + half*8
      float s = c[i] * 0.125f;                   // 1/sqrt(64)
      if (pmask || (diag && l15 > row)) s = NEGINF;
      mrow[i] = fmaxf(mrow[i], s);
    }
  }
#pragma unroll
  for (int m = 1; m < 16; m <<= 1)
#pragma unroll
    for (int i = 0; i < 8; ++i)
      mrow[i] = fmaxf(mrow[i], __shfl_xor(mrow[i], m, 32));
  // mrow[i] = full row max of row (half*8 + i) on every lane.

  if (l15 == 0) {
#pragma unroll
    for (int i = 0; i < 8; ++i) rowmax[half * 8 + i] = mrow[i];
  }

  // ===== Sweep 2: recompute scores, exp, f16 probs to LDS, row sums =====
  float srow[8];
#pragma unroll
  for (int i = 0; i < 8; ++i) srow[i] = 0.0f;

  for (int kt = 0; kt < ktiles; ++kt) {
    v16h b0, b1;
    loadKfrag(kt, &b0, &b1);
    v8f c = {};
    c = WMMA_F16(a0, b0, c);
    c = WMMA_F16(a1, b1, c);
    const bool pmask = (pm[kt * 16 + l15] == 1);
    const bool diag  = (kt == qt);
    const int  key   = kt * 16 + l15;
#pragma unroll
    for (int i = 0; i < 8; ++i) {
      const int row = half * 8 + i;
      float s = c[i] * 0.125f;
      if (pmask || (diag && l15 > row)) s = NEGINF;
      const float p = __expf(s - mrow[i]);       // p in (0,1], f16-safe
      srow[i] += p;
      probs[row * SEQ + key] = (_Float16)p;
    }
  }
#pragma unroll
  for (int m = 1; m < 16; m <<= 1)
#pragma unroll
    for (int i = 0; i < 8; ++i)
      srow[i] += __shfl_xor(srow[i], m, 32);

  // Analytic tail: columns [16*ktiles, SEQ) all hold exp(-1e9 - m_row).
  const int tailStart = ktiles * 16;
  const int tailCnt   = SEQ - tailStart;
#pragma unroll
  for (int i = 0; i < 8; ++i)
    srow[i] += (float)tailCnt * __expf(NEGINF - mrow[i]);
  // srow[i] now = full softmax denominator of row (half*8 + i).

  if (l15 == 0) {
#pragma unroll
    for (int i = 0; i < 8; ++i) rowsum[half * 8 + i] = srow[i];
  }
  __syncthreads();

  // ---- Fill prob tail (lane = row order; needs staged rowmax) ----
  if (tailCnt > 0) {
    const _Float16 pt = (_Float16)__expf(NEGINF - rowmax[l15]);
    v8h ptv;
#pragma unroll
    for (int e = 0; e < 8; ++e) ptv[e] = pt;
    const int thalf = tailCnt >> 1;              // multiple of 8
    _Float16* rowp = probs + l15 * SEQ + tailStart + half * thalf;
    for (int c = 0; c < thalf; c += 8)
      *(v8h*)(rowp + c) = ptv;                   // ds_store_b128
  }
  __syncthreads();

  // ===== Write normalized attention weights (the 1 GiB HBM stream) =====
  float* attnp = Attn + (size_t)bh * SEQ * SEQ + (size_t)q0 * SEQ;
  for (int r = 0; r < 16; ++r) {
    const float rinv = 1.0f / rowsum[r];
    const _Float16* src = probs + r * SEQ;
    float* dst = attnp + (size_t)r * SEQ;
    for (int c0 = lane * 8; c0 < SEQ; c0 += 32 * 8) {
      v8h ph = *(const v8h*)(src + c0);          // ds_load_b128
      v4f lo, hi;
#pragma unroll
      for (int e = 0; e < 4; ++e) {
        lo[e] = (float)ph[e] * rinv;
        hi[e] = (float)ph[e + 4] * rinv;
      }
      *(v4f*)(dst + c0)     = lo;                // global_store_b128
      *(v4f*)(dst + c0 + 4) = hi;
    }
  }

  // ================= P * V via WMMA =================
  v8f o0 = {}, o1 = {}, o2 = {}, o3 = {};
  for (int kc = 0; kc < SEQ / 32; ++kc) {
    v16h pa;  // A = probs[16 rows][32 keys]: two contiguous 16B runs per lane
    {
      const _Float16* pr = probs + l15 * SEQ + kc * 32 + half * 8;
      v8h plo = *(const v8h*)pr;                 // ds_load_b128
      v8h phi = *(const v8h*)(pr + 16);
#pragma unroll
      for (int e = 0; e < 8; ++e) { pa[e] = plo[e]; pa[e + 8] = phi[e]; }
    }
    // B = V[32 keys][16 dh cols] per dh-group; lane col = l15, K = half*8+e+(e&8)
    const float* vb = vh + (size_t)(kc * 32 + half * 8) * DHEAD + l15;
    v16h vb0, vb1, vb2, vb3;
#pragma unroll
    for (int e = 0; e < 16; ++e) {
      const float* vp = vb + (size_t)(e + (e & 8)) * DHEAD;
      vb0[e] = (_Float16)vp[0];
      vb1[e] = (_Float16)vp[16];
      vb2[e] = (_Float16)vp[32];
      vb3[e] = (_Float16)vp[48];
    }
    o0 = WMMA_F16(pa, vb0, o0);
    o1 = WMMA_F16(pa, vb1, o1);
    o2 = WMMA_F16(pa, vb2, o2);
    o3 = WMMA_F16(pa, vb3, o3);
  }

  // ===== Normalize O rows and store =====
  float* op = Out + (size_t)bh * SEQ * DHEAD + (size_t)q0 * DHEAD;
#pragma unroll
  for (int i = 0; i < 8; ++i) {
    const int row = half * 8 + i;                // D layout row
    const float rinv = 1.0f / srow[i];
    float* orow = op + (size_t)row * DHEAD + l15;
    orow[0]  = o0[i] * rinv;
    orow[16] = o1[i] * rinv;
    orow[32] = o2[i] * rinv;
    orow[48] = o3[i] * rinv;
  }
}

extern "C" void kernel_launch(void* const* d_in, const int* in_sizes, int n_in,
                              void* d_out, int out_size, void* d_ws, size_t ws_size,
                              hipStream_t stream) {
  (void)in_sizes; (void)n_in; (void)out_size; (void)d_ws; (void)ws_size;
  const float* q   = (const float*)d_in[0];
  const float* k   = (const float*)d_in[1];
  const float* v   = (const float*)d_in[2];
  // d_in[3] = casual_mask [1,1,S,S]: reproduced analytically (col > row)
  const int* pad   = (const int*)d_in[4];

  float* out  = (float*)d_out;                                   // [B,H,S,Dh]
  float* attn = out + (size_t)NBATCH * NHEAD * SEQ * DHEAD;      // [B,H,S,S]

  const int blocks = NBATCH * NHEAD * (SEQ / 16);                // 8192
  const size_t lds = (size_t)16 * SEQ * sizeof(short) + 32 * sizeof(float);
  attn_fused_wmma<<<blocks, 32, lds, stream>>>(q, k, v, pad, out, attn);
}
